// NGPRadianceField_56135222559033
// MI455X (gfx1250) — matrix operand order repaired and checked by reading, not compiled, not run on hardware
//
#include <hip/hip_runtime.h>
#include <hip/hip_bf16.h>
#include <math.h>

// ---------------------------------------------------------------------------
// Instant-NGP radiance field forward for gfx1250 (MI455X), wave32 + WMMA.
// Each wave processes a 16-point tile; all GEMMs use v_wmma_f32_16x16x32_f16.
// Weights are stored FRAGMENT-MAJOR in LDS so a B fragment is one 32-byte
// contiguous load per lane (2x ds_load_b128) instead of 16 strided u16 loads.
// ---------------------------------------------------------------------------

typedef __attribute__((ext_vector_type(16))) _Float16 v16h;
typedef __attribute__((ext_vector_type(8)))  _Float16 v8h;
typedef __attribute__((ext_vector_type(8)))  float    v8f;

#define NGP_N_LEVELS 16
#define NGP_LOG2_T   19
#define NGP_T        (1u << NGP_LOG2_T)

__device__ __forceinline__ v8f wmma16(v16h a, v16h b, v8f c) {
    // D = A(16x32 f16) * B(32x16 f16) + C(16x16 f32)
    return __builtin_amdgcn_wmma_f32_16x16x32_f16(
        /*neg_a=*/false, a, /*neg_b=*/false, b,
        /*c_mod=*/(short)0, c, /*reuse_a=*/false, /*reuse_b=*/false);
}

// ---------------------------------------------------------------------------
// Fragment-major weight storage.
// A 32(K) x 16(N) f16 tile occupies 512 halves: lane l owns 16 contiguous
// halves at tile*512 + l*16, where element k of lane l corresponds to
// W[kc*32 + (l>>4)*8 + (k&7) + (k>>3)*16][nc*16 + (l&15)]  (the exact HW B-fragment layout).
// ---------------------------------------------------------------------------
__device__ __forceinline__ void fill_frags(_Float16* dst, const float* src,
                                           int K, int Ncols,
                                           int srcRows, int srcCols, int tid) {
    const int nN    = Ncols >> 4;
    const int total = K * Ncols;
    for (int i = tid; i < total; i += 256) {
        int tile = i >> 9;          // /512
        int rem  = i & 511;
        int l    = rem >> 4;        // owning lane
        int k    = rem & 15;
        int kc   = tile / nN;
        int nc   = tile - kc * nN;
        int row  = kc * 32 + (l >> 4) * 8 + (k & 7) + (k >> 3) * 16;
        int col  = nc * 16 + (l & 15);
        float v  = (row < srcRows && col < srcCols) ? src[row * srcCols + col]
                                                    : 0.0f;
        dst[i] = (_Float16)v;
    }
}

// B fragment: one 32-byte contiguous vector load per lane.
__device__ __forceinline__ v16h load_b_frag(const _Float16* frags,
                                            int kc, int nc, int nN, int lane) {
    const _Float16* src = frags + (((kc * nN + nc) << 9) + (lane << 4));
    return *(const v16h*)src;
}

// A-matrix fragment (16x32 f16) from row-major f16 staging [16][W].
// HW layout: lanes 0-15 (M=lane) hold K={kb..kb+7, kb+16..kb+23};
// lanes 16-31 hold K={kb+8..kb+15, kb+24..kb+31}: two contiguous 8-half runs.
__device__ __forceinline__ v16h load_a_frag(const _Float16* buf, int W,
                                            int kbase, int laneHi, int m) {
    const _Float16* r = buf + m * W + kbase + laneHi * 8;
    v8h lo = *(const v8h*)(r);
    v8h hi = *(const v8h*)(r + 16);
    v16h a;
#pragma unroll
    for (int j = 0; j < 8; ++j) { a[j] = lo[j]; a[j + 8] = hi[j]; }
    return a;
}

// Store C/D fragment (16x16 f32) with ReLU as f16 into [16][W] at col nbase.
// Layout: VGPR r -> (M = (lane>>4)*8 + r, N = lane&15).
__device__ __forceinline__ void store_relu_h(_Float16* buf, int W, int nbase,
                                             v8f d, int lane) {
    int n   = lane & 15;
    int mhi = (lane >> 4) * 8;
#pragma unroll
    for (int r = 0; r < 8; ++r) {
        float v = d[r];
        v = v > 0.0f ? v : 0.0f;
        buf[(mhi + r) * W + nbase + n] = (_Float16)v;
    }
}

__global__ __launch_bounds__(256) void ngp_forward(
    const float* __restrict__ positions,   // [N,3]
    const float* __restrict__ directions,  // [N,3]
    const float* __restrict__ aabb,        // [6]
    const float* __restrict__ table,       // [16, 2^19, 2]
    const float* __restrict__ w_base1,     // [32,64]
    const float* __restrict__ w_base2,     // [64,16]
    const float* __restrict__ w_head1,     // [31,64]
    const float* __restrict__ w_head2,     // [64,64]
    const float* __restrict__ w_head3,     // [64,3]
    float* __restrict__ out,               // rgb [N,3] then density [N]
    int N)
{
    // f16 weights in LDS, fragment-major (padded to WMMA-friendly shapes)
    __shared__ __align__(32) _Float16 s_wb1[32 * 64];  // 4 frags (kc=0, nc=0..3)
    __shared__ __align__(32) _Float16 s_wb2[64 * 16];  // 2 frags (kc=0..1)
    __shared__ __align__(32) _Float16 s_wh1[32 * 64];  // 4 frags, row 31 zero
    __shared__ __align__(32) _Float16 s_wh2[64 * 64];  // 8 frags
    __shared__ __align__(32) _Float16 s_wh3[64 * 16];  // 2 frags, cols 3..15 zero
    // per-wave activation staging (ping-pong), row-major [16][64]
    __shared__ __align__(32) _Float16 s_stagA[8][16 * 64];
    __shared__ __align__(32) _Float16 s_stagB[8][16 * 64];

    const int tid = threadIdx.x;

    fill_frags(s_wb1, w_base1, 32, 64, 32, 64, tid);
    fill_frags(s_wb2, w_base2, 64, 16, 64, 16, tid);
    fill_frags(s_wh1, w_head1, 32, 64, 31, 64, tid);   // pad row 31 -> 0
    fill_frags(s_wh2, w_head2, 64, 64, 64, 64, tid);
    fill_frags(s_wh3, w_head3, 64, 16, 64, 3,  tid);   // pad cols 3..15 -> 0
    __syncthreads();

    const int lane   = tid & 31;
    const int wave   = tid >> 5;
    const int laneHi = lane >> 4;
    const int m      = lane & 15;
    const int pbase  = blockIdx.x * 128 + wave * 16;
    const int p      = pbase + m;
    const int pc     = p < N ? p : (N - 1);

    _Float16* stA = s_stagA[wave];
    _Float16* stB = s_stagB[wave];

    const float ax = aabb[0], ay = aabb[1], az = aabb[2];
    const float px = (positions[pc * 3 + 0] - ax) / (aabb[3] - ax);
    const float py = (positions[pc * 3 + 1] - ay) / (aabb[4] - ay);
    const float pz = (positions[pc * 3 + 2] - az) / (aabb[5] - az);

    // ---------------- Hash-grid encoding, direct to A-fragment ------------
    // Lane with laneHi=0 computes levels {0..3, 8..11}; laneHi=1 computes
    // {4..7, 12..15}: exactly the K slots this lane owns in the A layout.
    v16h a_enc = {};
#pragma unroll
    for (int t = 0; t < 8; ++t) {
        int level   = (t & 3) + 8 * (t >> 2) + 4 * laneHi;
        float scale = exp2f(4.0f + (float)level * (8.0f / 15.0f)) - 1.0f;
        int res     = (int)ceilf(scale) + 1;
        unsigned ur = (unsigned)res;
        bool use_hash = res > 80;                  // 80^3 <= 2^19 < 81^3

        float xs = px * scale + 0.5f;
        float ys = py * scale + 0.5f;
        float zs = pz * scale + 0.5f;
        float fx = floorf(xs), fy = floorf(ys), fz = floorf(zs);
        float wx = xs - fx, wy = ys - fy, wz = zs - fz;
        unsigned x0 = (unsigned)fx, y0 = (unsigned)fy, z0 = (unsigned)fz;

        const float* tl = table + (size_t)level * (size_t)(NGP_T * 2);
        float f0 = 0.0f, f1 = 0.0f;
#pragma unroll
        for (int c = 0; c < 8; ++c) {
            unsigned cx = x0 + (c & 1);
            unsigned cy = y0 + ((c >> 1) & 1);
            unsigned cz = z0 + ((c >> 2) & 1);
            unsigned ih = (cx * 1u ^ cy * 2654435761u ^ cz * 805459861u)
                          & (NGP_T - 1u);
            unsigned il = cx + cy * ur + cz * ur * ur;
            unsigned idx = use_hash ? ih : il;
            float wgt = ((c & 1)        ? wx : 1.0f - wx)
                      * (((c >> 1) & 1) ? wy : 1.0f - wy)
                      * (((c >> 2) & 1) ? wz : 1.0f - wz);
            f0 += tl[2u * idx + 0u] * wgt;
            f1 += tl[2u * idx + 1u] * wgt;
        }
        a_enc[2 * t + 0] = (_Float16)f0;
        a_enc[2 * t + 1] = (_Float16)f1;
    }

    // ---------------- Base layer 1: enc(16x32) @ wb1(32x64), ReLU ---------
#pragma unroll
    for (int tN = 0; tN < 4; ++tN) {
        v16h b  = load_b_frag(s_wb1, 0, tN, 4, lane);
        v8f acc = {};
        acc = wmma16(a_enc, b, acc);
        store_relu_h(stA, 64, tN * 16, acc, lane);
    }
    __syncthreads();

    // ---------------- Base layer 2: h(16x64) @ wb2(64x16) -----------------
    v8f bo = {};
    {
        v16h a0 = load_a_frag(stA, 64, 0,  laneHi, m);
        v16h a1 = load_a_frag(stA, 64, 32, laneHi, m);
        v16h b0 = load_b_frag(s_wb2, 0, 0, 1, lane);
        v16h b1 = load_b_frag(s_wb2, 1, 0, 1, lane);
        bo = wmma16(a0, b0, bo);
        bo = wmma16(a1, b1, bo);
    }

    // density out (col 0) + geo features (cols 1..15) into hh buffer (W=32)
    {
        int n   = lane & 15;
        int mhi = laneHi * 8;
#pragma unroll
        for (int r = 0; r < 8; ++r) {
            float v = bo[r];
            int mm  = mhi + r;
            int pp  = pbase + mm;
            if (n == 0) {
                if (pp < N) out[(size_t)3 * N + pp] = expf(v - 1.0f);
                stB[mm * 32 + 31] = (_Float16)0.0f;   // pad K=31
            } else {
                stB[mm * 32 + 16 + (n - 1)] = (_Float16)v;
            }
        }
    }

    // ---------------- SH degree-4 encode into hh[:,0..15] -----------------
    {
        float x = directions[pc * 3 + 0];
        float y = directions[pc * 3 + 1];
        float z = directions[pc * 3 + 2];
        float x2 = x * x, y2 = y * y, z2 = z * z;
        float xy = x * y, xz = x * z, yz = y * z;
        _Float16* hh = stB + m * 32;
        // lanes m and m+16 write identical values: benign duplicate stores
        hh[0]  = (_Float16)(0.28209479177387814f);
        hh[1]  = (_Float16)(-0.48860251190291987f * y);
        hh[2]  = (_Float16)(0.48860251190291987f * z);
        hh[3]  = (_Float16)(-0.48860251190291987f * x);
        hh[4]  = (_Float16)(1.0925484305920792f * xy);
        hh[5]  = (_Float16)(-1.0925484305920792f * yz);
        hh[6]  = (_Float16)(0.94617469575756f * z2 - 0.31539156525252f);
        hh[7]  = (_Float16)(-1.0925484305920792f * xz);
        hh[8]  = (_Float16)(0.5462742152960396f * (x2 - y2));
        hh[9]  = (_Float16)(0.5900435899266435f * y * (-3.0f * x2 + y2));
        hh[10] = (_Float16)(2.890611442640554f * xy * z);
        hh[11] = (_Float16)(0.4570457994644657f * y * (1.0f - 5.0f * z2));
        hh[12] = (_Float16)(0.3731763325901154f * z * (5.0f * z2 - 3.0f));
        hh[13] = (_Float16)(0.4570457994644657f * x * (1.0f - 5.0f * z2));
        hh[14] = (_Float16)(1.445305721320277f * z * (x2 - y2));
        hh[15] = (_Float16)(0.5900435899266435f * x * (x2 - 3.0f * y2));
    }
    __syncthreads();

    // ---------------- Head layer 1: hh(16x32) @ wh1(32x64), ReLU ----------
    {
        v16h ah = load_a_frag(stB, 32, 0, laneHi, m);
#pragma unroll
        for (int tN = 0; tN < 4; ++tN) {
            v16h b  = load_b_frag(s_wh1, 0, tN, 4, lane);
            v8f acc = {};
            acc = wmma16(ah, b, acc);
            store_relu_h(stA, 64, tN * 16, acc, lane);
        }
    }
    __syncthreads();

    // ---------------- Head layer 2: h(16x64) @ wh2(64x64), ReLU -----------
    {
        v16h a0 = load_a_frag(stA, 64, 0,  laneHi, m);
        v16h a1 = load_a_frag(stA, 64, 32, laneHi, m);
#pragma unroll
        for (int tN = 0; tN < 4; ++tN) {
            v16h b0 = load_b_frag(s_wh2, 0, tN, 4, lane);
            v16h b1 = load_b_frag(s_wh2, 1, tN, 4, lane);
            v8f acc = {};
            acc = wmma16(a0, b0, acc);
            acc = wmma16(a1, b1, acc);
            store_relu_h(stB, 64, tN * 16, acc, lane);
        }
    }
    __syncthreads();

    // ---------------- Head layer 3: h(16x64) @ wh3(64x16), sigmoid --------
    {
        v16h a0 = load_a_frag(stB, 64, 0,  laneHi, m);
        v16h a1 = load_a_frag(stB, 64, 32, laneHi, m);
        v16h b0 = load_b_frag(s_wh3, 0, 0, 1, lane);
        v16h b1 = load_b_frag(s_wh3, 1, 0, 1, lane);
        v8f acc = {};
        acc = wmma16(a0, b0, acc);
        acc = wmma16(a1, b1, acc);

        int n = lane & 15;
        if (n < 3) {
#pragma unroll
            for (int r = 0; r < 8; ++r) {
                int mm = laneHi * 8 + r;
                int pp = pbase + mm;
                if (pp < N) {
                    float v = acc[r];
                    out[(size_t)pp * 3 + n] = 1.0f / (1.0f + expf(-v));
                }
            }
        }
    }
}

extern "C" void kernel_launch(void* const* d_in, const int* in_sizes, int n_in,
                              void* d_out, int out_size, void* d_ws, size_t ws_size,
                              hipStream_t stream) {
    const float* positions  = (const float*)d_in[0];
    const float* directions = (const float*)d_in[1];
    const float* aabb       = (const float*)d_in[2];
    const float* table      = (const float*)d_in[3];
    const float* w_base1    = (const float*)d_in[4];
    const float* w_base2    = (const float*)d_in[5];
    const float* w_head1    = (const float*)d_in[6];
    const float* w_head2    = (const float*)d_in[7];
    const float* w_head3    = (const float*)d_in[8];
    float* out = (float*)d_out;

    int N = in_sizes[0] / 3;
    int blocks = (N + 127) / 128;   // 128 points per block (8 waves x 16)
    ngp_forward<<<blocks, 256, 0, stream>>>(
        positions, directions, aabb, table,
        w_base1, w_base2, w_head1, w_head2, w_head3, out, N);
}
